// CausalSelfAttention_9388798509358
// MI455X (gfx1250) — matrix operand order repaired and checked by reading, not compiled, not run on hardware
//
#include <hip/hip_runtime.h>

typedef __attribute__((ext_vector_type(16))) _Float16 v16h;
typedef __attribute__((ext_vector_type(8)))  float    v8f;

static constexpr int Bc = 2, Sc = 2048, Dc = 1024, Hc = 16, DKc = 64;

union FragH {
    v16h h;
    uint4 u[2];
};

union H8 {
    _Float16 h[8];
    uint4 u;
};

// ---------------------------------------------------------------------------
// Kernel 1: fused QKV projection.  out = x @ W^T + b, mixed precision,
// register-double-buffered global->LDS staging.
//   grid = (N/128, M/128, 3)  block = 256 (8 waves, 2x4 wave grid)
//   z=0 -> Q (scaled by 1/sqrt(DK)) stored [b,h,s,dk] f16
//   z=1 -> K stored [b,h,s,dk] f16
//   z=2 -> V stored transposed [b,h,dk,s] f16 (contiguous PV B-fragments)
// ---------------------------------------------------------------------------
__global__ __launch_bounds__(256) void qkv_proj_kernel(
    const float* __restrict__ x,
    const float* __restrict__ Wq, const float* __restrict__ bq,
    const float* __restrict__ Wk, const float* __restrict__ bk,
    const float* __restrict__ Wv, const float* __restrict__ bv,
    _Float16* __restrict__ qh, _Float16* __restrict__ kh,
    _Float16* __restrict__ vT)
{
    const int z = blockIdx.z;
    const float* __restrict__ W    = (z == 0) ? Wq : (z == 1) ? Wk : Wv;
    const float* __restrict__ bias = (z == 0) ? bq : (z == 1) ? bk : bv;

    const int n0 = blockIdx.x * 128;
    const int m0 = blockIdx.y * 128;

    // LDS tiles, padded to 40 halfs per row (80B: conflict-free b128 reads)
    __shared__ _Float16 As[128 * 40];   // x tile   [m][k]
    __shared__ _Float16 Bs[128 * 40];   // W tile   [n][k]

    const int tid  = threadIdx.x;
    const int wv   = tid >> 5;
    const int lane = tid & 31;
    const int hf   = lane >> 4;      // half-wave select
    const int ln   = lane & 15;
    const int wm   = wv >> 2;        // 0..1  (64 rows each)
    const int wn   = wv & 3;         // 0..3  (32 cols each)

    v8f acc[4][2];
    for (int a = 0; a < 4; ++a)
        for (int b2 = 0; b2 < 2; ++b2)
            for (int e = 0; e < 8; ++e) acc[a][b2][e] = 0.0f;

    const int rstage = tid >> 3;          // 0..31
    const int cstage = (tid & 7) * 4;     // 0,4,...,28

    // register staging buffers for the global->LDS pipeline
    float4 ax[4], wx[4];
    const size_t xbase[4] = {
        (size_t)(m0 + 0  + rstage) * Dc + cstage,
        (size_t)(m0 + 32 + rstage) * Dc + cstage,
        (size_t)(m0 + 64 + rstage) * Dc + cstage,
        (size_t)(m0 + 96 + rstage) * Dc + cstage };
    const size_t wbase[4] = {
        (size_t)(n0 + 0  + rstage) * Dc + cstage,
        (size_t)(n0 + 32 + rstage) * Dc + cstage,
        (size_t)(n0 + 64 + rstage) * Dc + cstage,
        (size_t)(n0 + 96 + rstage) * Dc + cstage };

    auto gload = [&](int k0) {
        for (int rr = 0; rr < 4; ++rr) {
            ax[rr] = *reinterpret_cast<const float4*>(&x[xbase[rr] + k0]);
            wx[rr] = *reinterpret_cast<const float4*>(&W[wbase[rr] + k0]);
        }
    };

    gload(0);
    constexpr int NK = Dc / 32;   // 32 K-tiles
    for (int kt = 0; kt < NK; ++kt) {
        // write staged registers (tile kt) to LDS with f32 -> f16 conversion
        for (int rr = 0; rr < 4; ++rr) {
            const int row = rr * 32 + rstage;
            As[row * 40 + cstage + 0] = (_Float16)ax[rr].x;
            As[row * 40 + cstage + 1] = (_Float16)ax[rr].y;
            As[row * 40 + cstage + 2] = (_Float16)ax[rr].z;
            As[row * 40 + cstage + 3] = (_Float16)ax[rr].w;
            Bs[row * 40 + cstage + 0] = (_Float16)wx[rr].x;
            Bs[row * 40 + cstage + 1] = (_Float16)wx[rr].y;
            Bs[row * 40 + cstage + 2] = (_Float16)wx[rr].z;
            Bs[row * 40 + cstage + 3] = (_Float16)wx[rr].w;
        }
        __syncthreads();

        // issue next tile's global loads; latency hidden behind WMMA compute
        if (kt + 1 < NK) gload((kt + 1) * 32);

        // B fragments: lane holds column n, 16 contiguous K
        FragH fb[2];
        for (int nt = 0; nt < 2; ++nt) {
            const int rowB = wn * 32 + nt * 16 + ln;
            const int offB = hf ? 16 : 0;
            fb[nt].u[0] = *reinterpret_cast<const uint4*>(&Bs[rowB * 40 + offB]);
            fb[nt].u[1] = *reinterpret_cast<const uint4*>(&Bs[rowB * 40 + offB + 8]);
        }
        for (int mt = 0; mt < 4; ++mt) {
            // A fragment: lane holds row m; lo half K{0..15}, hi K{16..31}
            FragH fa;
            const int rowA = wm * 64 + mt * 16 + ln;
            const int offA = hf ? 16 : 0;
            fa.u[0] = *reinterpret_cast<const uint4*>(&As[rowA * 40 + offA]);
            fa.u[1] = *reinterpret_cast<const uint4*>(&As[rowA * 40 + offA + 8]);
            for (int nt = 0; nt < 2; ++nt) {
                acc[mt][nt] = __builtin_amdgcn_wmma_f32_16x16x32_f16(
                    false, fa.h, false, fb[nt].h, (short)0, acc[mt][nt],
                    false, false);
            }
        }
        __syncthreads();
    }

    // --- epilogue: bias add, (Q scale), head-split store -------------------
    // A 128-row M tile never crosses a batch boundary (S % 128 == 0).
    const int bb  = m0 >> 11;                         // uniform batch
    const int ssb = (m0 & 2047) + wm * 64 + hf * 8;   // base seq index

    for (int nt = 0; nt < 2; ++nt) {
        const int n    = n0 + wn * 32 + nt * 16 + ln;
        const float bval = bias[n];
        const int hh = n >> 6;       // head
        const int dk = n & 63;       // dim within head
        const size_t bh = (size_t)(bb * Hc + hh);

        if (z == 2) {
            // V^T: consecutive i -> consecutive seq -> pack 8 halfs, b128 store
            _Float16* p = vT + (bh * DKc + dk) * Sc + ssb;
            for (int mt = 0; mt < 4; ++mt) {
                H8 pk;
                for (int i = 0; i < 8; ++i)
                    pk.h[i] = (_Float16)(acc[mt][nt][i] + bval);
                *reinterpret_cast<uint4*>(&p[mt * 16]) = pk.u;
            }
        } else {
            const float scale = (z == 0) ? 0.125f : 1.0f;   // 1/sqrt(DK) into Q
            _Float16* p = ((z == 0) ? qh : kh) + (bh * Sc + ssb) * DKc + dk;
            for (int mt = 0; mt < 4; ++mt)
                for (int i = 0; i < 8; ++i)
                    p[(size_t)(mt * 16 + i) * DKc] =
                        (_Float16)((acc[mt][nt][i] + bval) * scale);
        }
    }
}

// ---------------------------------------------------------------------------
// One 32-key flash-attention step for one wave (16 query rows).
// ---------------------------------------------------------------------------
template <bool MASK>
__device__ __forceinline__ void attn_step(
    const FragH (&qa)[2], const _Float16* __restrict__ kp,
    const _Float16* __restrict__ vp, _Float16* __restrict__ pw,
    float (&mrow)[8], float (&lrow)[8], v8f (&o)[4],
    int kb, int qlo, int hf, int ln)
{
    // ---- scores S = (Q/sqrt(dk)) @ K^T for 16 rows x 32 keys ----
    v8f s0, s1;
    for (int e = 0; e < 8; ++e) { s0[e] = 0.0f; s1[e] = 0.0f; }
    for (int ks = 0; ks < 2; ++ks) {
        const int dk0 = ks * 32 + (hf ? 16 : 0);
        FragH f0, f1;
        f0.u[0] = *reinterpret_cast<const uint4*>(&kp[(size_t)(kb + ln) * DKc + dk0]);
        f0.u[1] = *reinterpret_cast<const uint4*>(&kp[(size_t)(kb + ln) * DKc + dk0 + 8]);
        f1.u[0] = *reinterpret_cast<const uint4*>(&kp[(size_t)(kb + 16 + ln) * DKc + dk0]);
        f1.u[1] = *reinterpret_cast<const uint4*>(&kp[(size_t)(kb + 16 + ln) * DKc + dk0 + 8]);
        s0 = __builtin_amdgcn_wmma_f32_16x16x32_f16(false, qa[ks].h, false, f0.h,
                                                    (short)0, s0, false, false);
        s1 = __builtin_amdgcn_wmma_f32_16x16x32_f16(false, qa[ks].h, false, f1.h,
                                                    (short)0, s1, false, false);
    }

    // ---- (mask +) online softmax (rows live across 16-lane halves) ----
    for (int i = 0; i < 8; ++i) {
        if (MASK) {
            const int row = qlo + i + hf * 8;
            if (kb + ln > row)      s0[i] = -1e9f;
            if (kb + 16 + ln > row) s1[i] = -1e9f;
        }
        float tmax = fmaxf(s0[i], s1[i]);
        for (int d2 = 1; d2 < 16; d2 <<= 1)
            tmax = fmaxf(tmax, __shfl_xor(tmax, d2, 32));
        const float mn    = fmaxf(mrow[i], tmax);
        const float alpha = __expf(mrow[i] - mn);
        const float p0 = __expf(s0[i] - mn);
        const float p1 = __expf(s1[i] - mn);
        float rs = p0 + p1;
        for (int d2 = 1; d2 < 16; d2 <<= 1)
            rs += __shfl_xor(rs, d2, 32);
        lrow[i] = lrow[i] * alpha + rs;
        mrow[i] = mn;
        for (int t4 = 0; t4 < 4; ++t4) o[t4][i] *= alpha;
        const int lr = i + hf * 8;
        pw[lr * 40 + ln]      = (_Float16)p0;
        pw[lr * 40 + 16 + ln] = (_Float16)p1;
    }

    // ---- O += P @ V  (K = 32 keys; 4 output tiles of 16 dims) ----
    FragH pa;
    {
        const int offP = hf ? 16 : 0;
        pa.u[0] = *reinterpret_cast<const uint4*>(&pw[ln * 40 + offP]);
        pa.u[1] = *reinterpret_cast<const uint4*>(&pw[ln * 40 + offP + 8]);
    }
    for (int t4 = 0; t4 < 4; ++t4) {
        FragH fv;
        const int n   = t4 * 16 + ln;
        const int off = kb + (hf ? 16 : 0);
        fv.u[0] = *reinterpret_cast<const uint4*>(&vp[(size_t)n * Sc + off]);
        fv.u[1] = *reinterpret_cast<const uint4*>(&vp[(size_t)n * Sc + off + 8]);
        o[t4] = __builtin_amdgcn_wmma_f32_16x16x32_f16(false, pa.h, false, fv.h,
                                                       (short)0, o[t4], false, false);
    }
}

// ---------------------------------------------------------------------------
// Kernel 2: flash attention with online softmax.
//   grid = (S/128, B*H)  block = 256; each wave owns 16 query rows.
// ---------------------------------------------------------------------------
__global__ __launch_bounds__(256) void attn_kernel(
    const _Float16* __restrict__ qh, const _Float16* __restrict__ kh,
    const _Float16* __restrict__ vT, float* __restrict__ out)
{
    const int bh = blockIdx.y;
    const int bb = bh >> 4;
    const int hh = bh & 15;
    const int q0 = blockIdx.x * 128;

    const int tid  = threadIdx.x;
    const int wv   = tid >> 5;
    const int lane = tid & 31;
    const int hf   = lane >> 4;
    const int ln   = lane & 15;
    const int qlo  = q0 + wv * 16;

    const _Float16* __restrict__ qp = qh + (size_t)bh * Sc * DKc;
    const _Float16* __restrict__ kp = kh + (size_t)bh * Sc * DKc;
    const _Float16* __restrict__ vp = vT + (size_t)bh * DKc * Sc;

    // per-wave P relayout buffer (acc layout -> A-fragment layout), padded rows
    __shared__ _Float16 Pbuf[8 * 16 * 40];
    _Float16* pw = &Pbuf[wv * 16 * 40];

    // Q fragments for both K-steps of DK=64, held in registers for whole loop
    FragH qa[2];
    {
        const int qr = qlo + ln;
        for (int ks = 0; ks < 2; ++ks) {
            const int off = ks * 32 + (hf ? 16 : 0);
            qa[ks].u[0] = *reinterpret_cast<const uint4*>(&qp[(size_t)qr * DKc + off]);
            qa[ks].u[1] = *reinterpret_cast<const uint4*>(&qp[(size_t)qr * DKc + off + 8]);
        }
    }

    float mrow[8], lrow[8];
    v8f o[4];
    for (int i = 0; i < 8; ++i) { mrow[i] = -3.0e38f; lrow[i] = 0.0f; }
    for (int t4 = 0; t4 < 4; ++t4)
        for (int e = 0; e < 8; ++e) o[t4][e] = 0.0f;

    const int kend  = qlo + 16;  // keys needed: 0 .. qlo+15 (causal)
    // blocks with kb+31 <= qlo need no mask
    const int kfull = (qlo >= 31) ? ((((qlo - 31) >> 5) + 1) << 5) : 0;

    int kb = 0;
    for (; kb < kfull; kb += 32) {
        // prefetch next block's K rows and V rows (wave-uniform guard)
        if (kb + 32 < kend) {
            __builtin_prefetch(&kp[(size_t)(kb + 32 + lane) * DKc], 0, 3);
            __builtin_prefetch(&vp[(size_t)lane * Sc + kb + 32], 0, 3);
            __builtin_prefetch(&vp[(size_t)(32 + lane) * Sc + kb + 32], 0, 3);
        }
        attn_step<false>(qa, kp, vp, pw, mrow, lrow, o, kb, qlo, hf, ln);
    }
    for (; kb < kend; kb += 32) {
        attn_step<true>(qa, kp, vp, pw, mrow, lrow, o, kb, qlo, hf, ln);
    }

    // ---- epilogue: O / l, merge heads back to [b, s, d] f32 ----
    float* op = out + ((size_t)bb * Sc + qlo + hf * 8) * Dc + hh * DKc + ln;
    for (int i = 0; i < 8; ++i) {
        const float inv = 1.0f / lrow[i];
        for (int t4 = 0; t4 < 4; ++t4)
            op[(size_t)i * Dc + t4 * 16] = o[t4][i] * inv;
    }
}

// ---------------------------------------------------------------------------
extern "C" void kernel_launch(void* const* d_in, const int* in_sizes, int n_in,
                              void* d_out, int out_size, void* d_ws, size_t ws_size,
                              hipStream_t stream) {
    (void)in_sizes; (void)n_in; (void)out_size; (void)ws_size;
    const float* x  = (const float*)d_in[0];
    const float* Wq = (const float*)d_in[1];
    const float* bq = (const float*)d_in[2];
    const float* Wk = (const float*)d_in[3];
    const float* bk = (const float*)d_in[4];
    const float* Wv = (const float*)d_in[5];
    const float* bv = (const float*)d_in[6];
    float* out = (float*)d_out;

    const size_t nq = (size_t)Bc * Hc * Sc * DKc;   // 4,194,304 elems
    _Float16* qh = (_Float16*)d_ws;
    _Float16* kh = qh + nq;
    _Float16* vT = kh + nq;

    // Kernel 1: QKV projections. grid = (N/128, M/128, {Q,K,V})
    qkv_proj_kernel<<<dim3(Dc / 128, (Bc * Sc) / 128, 3), 256, 0, stream>>>(
        x, Wq, bq, Wk, bk, Wv, bv, qh, kh, vT);

    // Kernel 2: causal flash attention. grid = (S/128, B*H)
    attn_kernel<<<dim3(Sc / 128, Bc * Hc), 256, 0, stream>>>(qh, kh, vT, out);
}
